// RGLRU_87960930222745
// MI455X (gfx1250) — matrix operand order repaired and checked by reading, compiled, not run on hardware
//
#include <hip/hip_runtime.h>
#include <hip/hip_bf16.h>
#include <math.h>

typedef __bf16 v16bf __attribute__((ext_vector_type(16)));
typedef float  v8f   __attribute__((ext_vector_type(8)));

#define K_WIDTH 4096
#define NROWS   8192
#define NHEADS  8
#define BW      512
#define TCHUNK  128
#define NCHUNK  (NROWS / TCHUNK)   // 64
#define CTILE   64                 // channels per workgroup
#define JT_PER_HEAD (BW / 16)      // 32 N-tiles per head
#define KT_PER_HEAD (BW / 32)      // 16 K-tiles per head

// ---- workspace layout (bytes) ----
// wsB  : [2 gates][8 heads][32 jt][16 kt][32 lanes][16 bf16]  = 8 MB
// cumA : [8192][4096] float                                   = 128 MB
// hin  : [64 chunks][4096] float                              = 1 MB
// sp   : [4096] float  (softplus(a_param))                    = 16 KB
static const size_t WSB_BYTES  = (size_t)2 * NHEADS * JT_PER_HEAD * KT_PER_HEAD * 32 * 16 * 2;
static const size_t CUMA_OFF   = WSB_BYTES;
static const size_t CUMA_BYTES = (size_t)NROWS * K_WIDTH * 4;
static const size_t HIN_OFF    = CUMA_OFF + CUMA_BYTES;
static const size_t HIN_BYTES  = (size_t)NCHUNK * K_WIDTH * 4;
static const size_t SP_OFF     = HIN_OFF + HIN_BYTES;

// ---------------- Kernel 0a: swizzle fp32 weights into WMMA B-fragment bf16 layout ----------------
// B is KxN (32x16 per fragment). Lane L holds column N = L&15 of the 32-row K-tile;
// element e is K = (e<8 ? e : e+8) + (L>>4)*8 (matches the A fragment k order).
__global__ __launch_bounds__(256) void rglru_swizzle_w(const float* __restrict__ ig_w,
                                                       const float* __restrict__ ag_w,
                                                       __bf16* __restrict__ wsB) {
    unsigned gtid = blockIdx.x * blockDim.x + threadIdx.x;
    unsigned lane = gtid & 31u;
    unsigned rest = gtid >> 5;
    unsigned kt   = rest & 15u; rest >>= 4;
    unsigned jt   = rest & 31u; rest >>= 5;
    unsigned head = rest & 7u;
    unsigned g    = rest >> 3;
    if (g >= 2) return;
    const float* w = (g == 0) ? ig_w : ag_w;
    const float* src = w + (size_t)head * BW * BW + jt * 16 + (lane & 15u);
    v16bf frag;
#pragma unroll
    for (int e = 0; e < 16; ++e) {
        unsigned Kk = kt * 32 + (e < 8 ? e : e + 8) + (lane >> 4) * 8;
        frag[e] = (__bf16)src[(size_t)Kk * BW];
    }
    size_t off = ((((size_t)g * NHEADS + head) * JT_PER_HEAD + jt) * KT_PER_HEAD + kt) * 32 + lane;
    *(v16bf*)(wsB + off * 16) = frag;
}

// ---------------- Kernel 0b: precompute per-channel softplus(a_param) ----------------
__global__ __launch_bounds__(256) void rglru_prep_sp(const float* __restrict__ a_param,
                                                     float* __restrict__ sp) {
    int j = blockIdx.x * blockDim.x + threadIdx.x;
    if (j < K_WIDTH) sp[j] = log1pf(__expf(a_param[j]));
}

// ---------------- Kernel 1: fused WMMA gate-GEMMs + elementwise + chunk-local scan ----------------
__global__ __launch_bounds__(256) void rglru_gemm_scan(
    const float* __restrict__ x, const int* __restrict__ s,
    const float* __restrict__ sp_tab,
    const float* __restrict__ ig_b, const float* __restrict__ ag_b,
    const __bf16* __restrict__ wsB,
    float* __restrict__ out, float* __restrict__ cumA)
{
    __shared__ float a_s[TCHUNK][CTILE + 1];
    __shared__ float b_s[TCHUNK][CTILE + 1];

    const int chunk   = blockIdx.x;            // 0..63 time chunks
    const int colTile = blockIdx.y;            // 0..63 channel tiles
    const int col0    = colTile * CTILE;
    const int head    = col0 / BW;
    const int jt0     = (col0 - head * BW) / 16;   // N-tile base within head

    const int tid   = threadIdx.x;
    const int wave  = tid >> 5;                // 0..7 -> M-tile (16 rows each)
    const int lane  = tid & 31;
    const int laneM = lane & 15;
    const int laneG = lane >> 4;

    const int rowBase = chunk * TCHUNK + wave * 16;

    v8f accX[4], accA[4];
#pragma unroll
    for (int i = 0; i < 4; ++i) { accX[i] = v8f{}; accA[i] = v8f{}; }

    const __bf16* wsBX = wsB + ((size_t)(0 * NHEADS + head) * JT_PER_HEAD) * KT_PER_HEAD * 32 * 16;
    const __bf16* wsBA = wsB + ((size_t)(1 * NHEADS + head) * JT_PER_HEAD) * KT_PER_HEAD * 32 * 16;
    const float*  xrow = x + (size_t)(rowBase + laneM) * K_WIDTH + head * BW;

    for (int kt = 0; kt < KT_PER_HEAD; ++kt) {
        // A fragment (16x32 bf16, ISA layout): lane holds row M=laneM;
        // element e -> K = kt*32 + (e<8 ? e : e+8) + laneG*8  (consecutive pairs => b64/b128 loads)
        v16bf afrag;
#pragma unroll
        for (int p = 0; p < 8; ++p) {
            int koff = ((p < 4) ? 2 * p : 2 * p + 8) + laneG * 8;
            float2 v = *(const float2*)(xrow + kt * 32 + koff);
            afrag[2 * p]     = (__bf16)v.x;
            afrag[2 * p + 1] = (__bf16)v.y;
        }
#pragma unroll
        for (int nt = 0; nt < 4; ++nt) {
            size_t boff = (((size_t)(jt0 + nt) * KT_PER_HEAD + kt) * 32 + lane) * 16;
            v16bf bx = *(const v16bf*)(wsBX + boff);
            v16bf ba = *(const v16bf*)(wsBA + boff);
            accX[nt] = __builtin_amdgcn_wmma_f32_16x16x32_bf16(
                false, afrag, false, bx, (short)0, accX[nt], false, false);
            accA[nt] = __builtin_amdgcn_wmma_f32_16x16x32_bf16(
                false, afrag, false, ba, (short)0, accA[nt], false, false);
        }
    }

    // Per-row reset flags (independent of channel): hoisted out of the nt loop.
    bool reset[8];
#pragma unroll
    for (int r = 0; r < 8; ++r) {
        int row = rowBase + laneG * 8 + r;
        reset[r] = (row == 0) || (s[row] != s[row - 1]);
    }

    // Epilogue: C layout -> element r is row (wave*16 + laneG*8 + r), col (nt*16 + laneM)
#pragma unroll
    for (int nt = 0; nt < 4; ++nt) {
        int col = col0 + nt * 16 + laneM;
        float sp  = sp_tab[col];       // softplus(a_param), precomputed
        float bxb = ig_b[col];
        float bab = ag_b[col];
#pragma unroll
        for (int r = 0; r < 8; ++r) {
            int rowLocal = wave * 16 + laneG * 8 + r;
            int row = chunk * TCHUNK + rowLocal;
            float gx = __builtin_amdgcn_rcpf(1.0f + __expf(-(accX[nt][r] + bxb)));
            float ga = __builtin_amdgcn_rcpf(1.0f + __expf(-(accA[nt][r] + bab)));
            float av = __expf(-8.0f * ga * sp);
            // a^2 == exp(2*log_a); sqrt(1-a^2) unless at a segment reset
            float mult = reset[r] ? 1.0f : sqrtf(fmaxf(0.0f, 1.0f - av * av));
            float xv = x[(size_t)row * K_WIDTH + col];
            a_s[rowLocal][nt * 16 + laneM] = av;
            b_s[rowLocal][nt * 16 + laneM] = xv * gx * mult;
        }
    }
    __syncthreads();

    // Chunk-local serial scan: 64 lanes, one channel each, 128 time steps.
    if (tid < CTILE) {
        float h = 0.0f, cum = 1.0f;
        for (int t = 0; t < TCHUNK; ++t) {
            float av = a_s[t][tid];
            float bv = b_s[t][tid];
            h = fmaf(av, h, bv);
            cum *= av;
            size_t gi = (size_t)(chunk * TCHUNK + t) * K_WIDTH + col0 + tid;
            out[gi]  = h;     // local prefix (missing carry-in)
            cumA[gi] = cum;   // inclusive product within chunk
        }
    }
}

// ---------------- Kernel 2: carry scan over chunk aggregates ----------------
__global__ __launch_bounds__(256) void rglru_chunk_carry(const float* __restrict__ out,
                                                         const float* __restrict__ cumA,
                                                         float* __restrict__ hin) {
    int j = blockIdx.x * blockDim.x + threadIdx.x;
    if (j >= K_WIDTH) return;
    float h = 0.0f;
    for (int c = 0; c < NCHUNK; ++c) {
        hin[(size_t)c * K_WIDTH + j] = h;
        size_t gi = ((size_t)c * TCHUNK + TCHUNK - 1) * K_WIDTH + j;
        h = fmaf(cumA[gi], h, out[gi]);
    }
}

// ---------------- Kernel 3: fixup out += cumA * h_in ----------------
__global__ __launch_bounds__(256) void rglru_fixup(float* __restrict__ out,
                                                   const float* __restrict__ cumA,
                                                   const float* __restrict__ hin) {
    size_t idx = (size_t)blockIdx.x * blockDim.x + threadIdx.x;
    const size_t total4 = (size_t)NROWS * K_WIDTH / 4;
    if (idx >= total4) return;
    size_t row = idx / (K_WIDTH / 4);
    size_t c4  = idx % (K_WIDTH / 4);
    size_t chunk = row / TCHUNK;
    float4 o  = ((const float4*)out)[idx];
    float4 ca = ((const float4*)cumA)[idx];
    float4 hv = ((const float4*)hin)[chunk * (K_WIDTH / 4) + c4];
    o.x = fmaf(ca.x, hv.x, o.x);
    o.y = fmaf(ca.y, hv.y, o.y);
    o.z = fmaf(ca.z, hv.z, o.z);
    o.w = fmaf(ca.w, hv.w, o.w);
    ((float4*)out)[idx] = o;
}

extern "C" void kernel_launch(void* const* d_in, const int* in_sizes, int n_in,
                              void* d_out, int out_size, void* d_ws, size_t ws_size,
                              hipStream_t stream) {
    const float* x       = (const float*)d_in[0];
    const int*   s       = (const int*)  d_in[1];
    const float* a_param = (const float*)d_in[2];
    const float* ig_w    = (const float*)d_in[3];
    const float* ig_b    = (const float*)d_in[4];
    const float* ag_w    = (const float*)d_in[5];
    const float* ag_b    = (const float*)d_in[6];
    float* out = (float*)d_out;

    char*   ws   = (char*)d_ws;
    __bf16* wsB  = (__bf16*)ws;
    float*  cumA = (float*)(ws + CUMA_OFF);
    float*  hin  = (float*)(ws + HIN_OFF);
    float*  sp   = (float*)(ws + SP_OFF);

    // 0a) weight swizzle: 2*8*32*16*32 fragments, one per thread
    rglru_swizzle_w<<<(2 * NHEADS * JT_PER_HEAD * KT_PER_HEAD * 32) / 256, 256, 0, stream>>>(
        ig_w, ag_w, wsB);
    // 0b) softplus(a_param) table
    rglru_prep_sp<<<K_WIDTH / 256, 256, 0, stream>>>(a_param, sp);

    // 1) fused WMMA GEMMs + gate epilogue + chunk-local scan
    dim3 g1(NCHUNK, K_WIDTH / CTILE);
    rglru_gemm_scan<<<g1, 256, 0, stream>>>(x, s, sp, ig_b, ag_b, wsB, out, cumA);

    // 2) carry scan across 64 chunks per channel
    rglru_chunk_carry<<<K_WIDTH / 256, 256, 0, stream>>>(out, cumA, hin);

    // 3) fixup
    size_t total4 = (size_t)NROWS * K_WIDTH / 4;
    rglru_fixup<<<(unsigned)((total4 + 255) / 256), 256, 0, stream>>>(out, cumA, hin);
}
